// ALiBiMultiHeadAttention_11536282157740
// MI455X (gfx1250) — matrix operand order repaired and checked by reading, not compiled, not run on hardware
//
#include <hip/hip_runtime.h>

// ---------------------------------------------------------------------------
// CDNA5 (gfx1250) bf16 WMMA + TDM implementation of ALiBi causal MHA.
// B=2, S=2048, D=2048, H=16, hd=128.
//   kernel 0a: x  (f32) -> xbf  (bf16, [4096][2048])
//   kernel 0b: W  (f32) -> wT   (bf16, transposed [6144][2048])
//   kernel 1 : kq/v GEMM, TDM double-buffered LDS, bf16 in / bf16 out
//   kernel 2 : flash attention (ALiBi + causal), pipelined TDM staging
// ---------------------------------------------------------------------------

typedef __attribute__((ext_vector_type(16))) __bf16 v16bf;
typedef __attribute__((ext_vector_type(8)))  float  v8f;
typedef __attribute__((ext_vector_type(4)))  unsigned int u32x4;
typedef __attribute__((ext_vector_type(8)))  int    i32x8;
typedef __attribute__((ext_vector_type(4)))  int    i32x4;

union AFrag {               // 8 VGPRs: 16 bf16 values per lane
  v16bf        v;
  unsigned int d[8];
};

__device__ __forceinline__ unsigned short f2bf(float f) {
  unsigned int u = __float_as_uint(f);
  u += 0x7fffu + ((u >> 16) & 1u);            // round-to-nearest-even
  return (unsigned short)(u >> 16);
}
__device__ __forceinline__ unsigned int pack2bf(float a, float b) {
  return (unsigned int)f2bf(a) | ((unsigned int)f2bf(b) << 16);
}

// ---------------------------------------------------------------------------
// Tensor Data Mover: 2-D bf16 tile load, global -> padded LDS rows.
// D# layout per CDNA5 ISA.
//   tile_d0: elements per row (bf16), tile_d1: rows
//   row_stride: elements between rows in memory
//   pad codes: interval c -> 2^(c+1) DWORDs between pads; amount p -> p+1 DWORDs
// Uses the 6-arg clang-23 builtin: (u32x4, i32x8, i32x4, i32x4, i32x8, cpol)
// ---------------------------------------------------------------------------
__device__ __forceinline__ void tdm_load_2d(
    unsigned int lds_addr, const void* gptr,
    unsigned int tile_d0, unsigned int tile_d1, unsigned int row_stride,
    unsigned int pad_interval_code, unsigned int pad_amount_code) {
  unsigned long long ga = (unsigned long long)(size_t)gptr;
  u32x4 g0;
  g0.x = 1u;                                      // count=1 (valid user D#)
  g0.y = lds_addr;                                // LDS byte address
  g0.z = (unsigned int)ga;                        // global_addr[31:0]
  g0.w = ((unsigned int)(ga >> 32) & 0x01ffffffu) // global_addr[56:32]
         | 0x80000000u;                           // type=2
  i32x8 g1;
  g1[0] = (int)((pad_amount_code << 25) | (pad_interval_code << 22) |
                (1u << 20) | (1u << 16));         // pad_enable, data_size=2B
  g1[1] = (int)((tile_d0 & 0xffffu) << 16);       // tensor_dim0 = tile_d0
  g1[2] = (int)((tile_d0 >> 16) | ((tile_d1 & 0xffffu) << 16)); // tensor_dim1
  g1[3] = (int)((tile_d1 >> 16) | ((tile_d0 & 0xffffu) << 16)); // tile_dim0
  g1[4] = (int)(tile_d1 & 0xffffu);               // tile_dim1 (tile_dim2=0)
  g1[5] = (int)row_stride;                        // tensor_dim0_stride[31:0]
  g1[6] = 0;
  g1[7] = 0;
  i32x4 z4 = {0, 0, 0, 0};                        // groups 2/3: unused (2-D)
  i32x8 z8 = {0, 0, 0, 0, 0, 0, 0, 0};
  __builtin_amdgcn_tensor_load_to_lds(g0, g1, z4, z4, z8, 0);
}

#define GM 4096
#define GK 2048
#define GN 6144
#define KQW 4096              // width of bf16 K|Q buffer
#define AKS 40                // LDS K-stride (ushort): 64B data + 16B pad

// ---------------------------------------------------------------------------
// Kernel 0a: x f32 -> bf16 (straight copy/convert, 8 elems/thread)
// ---------------------------------------------------------------------------
__global__ __launch_bounds__(256) void cvt_x_bf16(const float* __restrict__ X,
                                                  unsigned short* __restrict__ Xb) {
  size_t t = (size_t)blockIdx.x * 256 + threadIdx.x;
  const float4 f0 = *(const float4*)(X + t * 8);
  const float4 f1 = *(const float4*)(X + t * 8 + 4);
  uint4 o;
  o.x = pack2bf(f0.x, f0.y); o.y = pack2bf(f0.z, f0.w);
  o.z = pack2bf(f1.x, f1.y); o.w = pack2bf(f1.z, f1.w);
  *(uint4*)(Xb + t * 8) = o;
}

// ---------------------------------------------------------------------------
// Kernel 0b: W[k][n] f32 -> wT[n][k] bf16 via 64x64 LDS tile transpose
// ---------------------------------------------------------------------------
#define TP 72
__global__ __launch_bounds__(256) void cvt_wT_bf16(const float* __restrict__ W,
                                                   unsigned short* __restrict__ WT) {
  __shared__ unsigned short Tl[64 * TP];   // [n_local][k_local]
  const int tid = threadIdx.x;
  const int bn = blockIdx.x * 64, bk = blockIdx.y * 64;
#pragma unroll
  for (int it = 0; it < 4; ++it) {
    int id = tid + 256 * it;
    int r = id >> 4, c4 = (id & 15) * 4;           // r: k-local, c4: n-local
    const float4 f = *(const float4*)(W + (size_t)(bk + r) * GN + bn + c4);
    Tl[(c4 + 0) * TP + r] = f2bf(f.x);
    Tl[(c4 + 1) * TP + r] = f2bf(f.y);
    Tl[(c4 + 2) * TP + r] = f2bf(f.z);
    Tl[(c4 + 3) * TP + r] = f2bf(f.w);
  }
  __syncthreads();
#pragma unroll
  for (int it = 0; it < 4; ++it) {
    int id = tid + 256 * it;
    int rn = id >> 4, ck = (id & 15) * 4;          // rn: n-local, ck: k-local
    uint2 d = *(const uint2*)&Tl[rn * TP + ck];
    *(uint2*)(WT + (size_t)(bn + rn) * GK + bk + ck) = d;
  }
}

// ---------------------------------------------------------------------------
// Kernel 1: [4096,6144] = Xb[4096,2048] * WT^T + bias   (bf16 in, bf16 out)
// 128x128 tile, 8 waves (2x4), K-step 32, TDM double-buffered LDS.
// Epilogue: cols <4096 -> kq bf16; cols >=4096 (V) -> vt transposed bf16.
// ---------------------------------------------------------------------------
__global__ __launch_bounds__(256) void kqv_gemm_bf16(
    const unsigned short* __restrict__ Xb, const unsigned short* __restrict__ WT,
    const float* __restrict__ bias, unsigned short* __restrict__ kq,
    unsigned short* __restrict__ vt) {
  __shared__ unsigned short As[2][128 * AKS];   // [m][k], 80B padded rows
  __shared__ unsigned short Bs[2][128 * AKS];   // [n][k]

  const int tid  = threadIdx.x;
  const int lane = tid & 31, wv = tid >> 5;
  const int half = lane >> 4, l16 = lane & 15;
  const int wm = wv & 1, wn = wv >> 1;          // 2 x 4 wave grid
  const int bm = blockIdx.y * 128, bn = blockIdx.x * 128;

  const unsigned int as0 = (unsigned int)(size_t)&As[0][0];
  const unsigned int bs0 = (unsigned int)(size_t)&Bs[0][0];
  const unsigned int tileB = 128 * AKS * 2;     // bytes per LDS buffer

  const v8f vzero = {};
  v8f acc[4][2];
#pragma unroll
  for (int mt = 0; mt < 4; ++mt)
#pragma unroll
    for (int nt = 0; nt < 2; ++nt) acc[mt][nt] = vzero;

  // prologue: TDM-stage tile kk=0 into buffer 0
  if (wv == 0) {
    tdm_load_2d(as0, Xb + (size_t)bm * GK, 32, 128, GK, 3, 3);
    tdm_load_2d(bs0, WT + (size_t)bn * GK, 32, 128, GK, 3, 3);
    __builtin_amdgcn_s_wait_tensorcnt(0);
  }
  __syncthreads();

  for (int kk = 0; kk < GK; kk += 32) {
    const int cur = (kk >> 5) & 1;
    const bool more = (kk + 32) < GK;
    // async TDM load of next K tile into the other buffer (overlaps WMMA)
    if (more && wv == 0) {
      tdm_load_2d(as0 + (cur ^ 1) * tileB, Xb + (size_t)bm * GK + kk + 32,
                  32, 128, GK, 3, 3);
      tdm_load_2d(bs0 + (cur ^ 1) * tileB, WT + (size_t)bn * GK + kk + 32,
                  32, 128, GK, 3, 3);
    }

    // A fragment: lane m = l16; K = 16*(v>=4) + 8*half + 2*(v&3) (+pair)
    AFrag a[4], b[2];
#pragma unroll
    for (int mt = 0; mt < 4; ++mt) {
      const unsigned short* p = &As[cur][(wm * 64 + mt * 16 + l16) * AKS];
#pragma unroll
      for (int v = 0; v < 8; ++v)
        a[mt].d[v] = *(const unsigned int*)&p[(v >= 4 ? 16 : 0) + half * 8 + (v & 3) * 2];
    }
    // B fragment: lane n = l16; K = 16*(v>=4) + 8*half + 2*(v&3) (+pair)
#pragma unroll
    for (int nt = 0; nt < 2; ++nt) {
      const unsigned short* p = &Bs[cur][(wn * 32 + nt * 16 + l16) * AKS];
#pragma unroll
      for (int v = 0; v < 8; ++v)
        b[nt].d[v] = *(const unsigned int*)&p[(v >= 4 ? 16 : 0) + half * 8 + (v & 3) * 2];
    }
#pragma unroll
    for (int mt = 0; mt < 4; ++mt)
#pragma unroll
      for (int nt = 0; nt < 2; ++nt)
        acc[mt][nt] = __builtin_amdgcn_wmma_f32_16x16x32_bf16(
            false, a[mt].v, false, b[nt].v, (short)0, acc[mt][nt], false, false);

    if (more) {
      if (wv == 0) __builtin_amdgcn_s_wait_tensorcnt(0);
      __syncthreads();
    }
  }

  // epilogue: C element (M = r + 8*half, N = l16), bf16 outputs
#pragma unroll
  for (int mt = 0; mt < 4; ++mt)
#pragma unroll
    for (int nt = 0; nt < 2; ++nt) {
      int gn = bn + wn * 32 + nt * 16 + l16;
      float bb = bias[gn];
      if (gn < 4096) {                       // K | Q region
#pragma unroll
        for (int r = 0; r < 8; ++r) {
          int gm = bm + wm * 64 + mt * 16 + r + 8 * half;
          kq[(size_t)gm * KQW + gn] = f2bf(acc[mt][nt][r] + bb);
        }
      } else {                               // V region -> transposed vt
        int hh = (gn - 4096) >> 7, dd = (gn - 4096) & 127;
#pragma unroll
        for (int r = 0; r < 8; ++r) {
          int gm = bm + wm * 64 + mt * 16 + r + 8 * half;
          int bI = gm >> 11, s = gm & 2047;
          vt[((size_t)(bI * 16 + hh) * 128 + dd) * 2048 + s] = f2bf(acc[mt][nt][r] + bb);
        }
      }
    }
}

// ---------------------------------------------------------------------------
// Kernel 2: flash attention with ALiBi bias + causal mask.
// Grid: 512 blocks = B(2)*H(16)*Sq-tiles(16); 256 threads = 8 waves,
// each wave owns 16 query rows. Three LDS buffers (K, V^T, P); TDM loads
// are software-pipelined: V(j) overlaps S-GEMM+softmax, K(j+1) overlaps PV.
// TDM completes in order per wave, so s_wait_tensorcnt<=1 proves V(j) landed.
// ---------------------------------------------------------------------------
#define PS 136   // padded LDS row stride (ushort): 256B data + 16B pad

__global__ __launch_bounds__(256) void alibi_fattn(
    const unsigned short* __restrict__ kq, const unsigned short* __restrict__ vt,
    float* __restrict__ out) {
  __shared__ unsigned short Kt[128 * PS];  // K tile  [key][hd]
  __shared__ unsigned short Vb[128 * PS];  // V^T tile [hd][key]
  __shared__ unsigned short Pt[128 * PS];  // probabilities bf16 [row][key]

  const int S = 2048;
  const int tid = threadIdx.x, lane = tid & 31, wv = tid >> 5;
  const int half = lane >> 4, l16 = lane & 15;
  const int z = blockIdx.x;
  const int qb = z & 15, h = (z >> 4) & 15, b = z >> 8;
  const float slope = exp2f(-0.5f * (float)(h + 1));   // 2^(-8(h+1)/16)
  const float scale = 0.08838834764831845f;            // 1/sqrt(128)

  const unsigned short* Kb = kq + (size_t)b * S * KQW + h * 128;        // K cols
  const unsigned short* Qb = Kb + 2048;                                  // Q cols
  const unsigned short* Vt = vt + (size_t)(b * 16 + h) * 128 * 2048;     // [hd][key]
  const int q0 = qb * 128 + wv * 16;                    // wave's first q row
  const unsigned int kt_lds = (unsigned int)(size_t)&Kt[0];
  const unsigned int vb_lds = (unsigned int)(size_t)&Vb[0];

  // Q A-fragments: direct bf16 dword loads (scale folded into score epilogue)
  AFrag qf[4];
#pragma unroll
  for (int c = 0; c < 4; ++c) {
    const unsigned short* qr = Qb + (size_t)(q0 + l16) * KQW + c * 32;
#pragma unroll
    for (int v = 0; v < 8; ++v) {
      int k = (v >= 4 ? 16 : 0) + half * 8 + (v & 3) * 2;
      qf[c].d[v] = *(const unsigned int*)&qr[k];
    }
  }

  const v8f vzero = {};
  v8f o[8];
  float mrun[8], lrun[8];
#pragma unroll
  for (int f = 0; f < 8; ++f) o[f] = vzero;
#pragma unroll
  for (int r = 0; r < 8; ++r) { mrun[r] = -__builtin_inff(); lrun[r] = 0.0f; }

  const int crow = tid >> 1, ccb = (tid & 1) * 64;   // prefetch geometry

  // prologue: kick off K(0) DMA
  if (wv == 0)
    tdm_load_2d(kt_lds, Kb, 128, 128, KQW, 5, 3);

  for (int j = 0; j <= qb; ++j) {      // causal: key blocks 0..qb
    // [A] K(j) complete & visible; prior PV reads of Vb finished
    if (wv == 0) __builtin_amdgcn_s_wait_tensorcnt(0);
    __syncthreads();
    if (wv == 0)                       // async V(j) DMA overlaps S + softmax
      tdm_load_2d(vb_lds, Vt + j * 128, 128, 128, 2048, 5, 3);
    if (j < qb) {  // global_prefetch_b8: warm L2 for next key block
      __builtin_prefetch(Kb + (size_t)((j + 1) * 128 + crow) * KQW + ccb, 0, 0);
      __builtin_prefetch(Vt + (size_t)crow * 2048 + (j + 1) * 128 + ccb, 0, 0);
    }

    // S = Q * K^T : 8 key-subtiles x 4 hd-chunks = 32 WMMA per wave
    v8f sf[8];
#pragma unroll
    for (int f = 0; f < 8; ++f) sf[f] = vzero;
#pragma unroll
    for (int c = 0; c < 4; ++c) {
#pragma unroll
      for (int f = 0; f < 8; ++f) {
        AFrag kf;
        const unsigned short* p = &Kt[(f * 16 + l16) * PS + c * 32];
#pragma unroll
        for (int v = 0; v < 8; ++v) kf.d[v] = *(const unsigned int*)&p[(v >= 4 ? 16 : 0) + half * 8 + (v & 3) * 2];
        sf[f] = __builtin_amdgcn_wmma_f32_16x16x32_bf16(
            false, qf[c].v, false, kf.v, (short)0, sf[f], false, false);
      }
    }

    // scale + ALiBi bias + causal mask on C-fragment coordinates
#pragma unroll
    for (int f = 0; f < 8; ++f) {
      int sk = j * 128 + f * 16 + l16;
#pragma unroll
      for (int r = 0; r < 8; ++r) {
        int sq = q0 + r + 8 * half;
        float sv = sf[f][r] * scale + slope * (float)(sk - sq);
        sf[f][r] = (sk <= sq) ? sv : -__builtin_inff();
      }
    }

    // online softmax (rows live in 16-lane half-groups; shfl masks 1..8)
    float pm[8];
#pragma unroll
    for (int r = 0; r < 8; ++r) {
      float m = sf[0][r];
#pragma unroll
      for (int f = 1; f < 8; ++f) m = fmaxf(m, sf[f][r]);
#pragma unroll
      for (int msk = 1; msk <= 8; msk <<= 1) m = fmaxf(m, __shfl_xor(m, msk, 32));
      pm[r] = m;
    }
    float corr[8];
#pragma unroll
    for (int r = 0; r < 8; ++r) {
      float nm = fmaxf(mrun[r], pm[r]);
      corr[r] = __expf(mrun[r] - nm);
      mrun[r] = nm;
    }
    float ps[8];
#pragma unroll
    for (int r = 0; r < 8; ++r) ps[r] = 0.0f;
#pragma unroll
    for (int f = 0; f < 8; ++f)
#pragma unroll
      for (int r = 0; r < 8; ++r) {
        float p = __expf(sf[f][r] - mrun[r]);   // masked -> exp(-inf) = 0
        sf[f][r] = p;
        ps[r] += p;
      }
#pragma unroll
    for (int r = 0; r < 8; ++r) {
#pragma unroll
      for (int msk = 1; msk <= 8; msk <<= 1) ps[r] += __shfl_xor(ps[r], msk, 32);
      lrun[r] = lrun[r] * corr[r] + ps[r];
    }
#pragma unroll
    for (int f = 0; f < 8; ++f)
#pragma unroll
      for (int r = 0; r < 8; ++r) o[f][r] *= corr[r];

    // P (bf16) -> LDS [row][key] (wave-local rows; read back by same wave)
#pragma unroll
    for (int f = 0; f < 8; ++f)
#pragma unroll
      for (int r = 0; r < 8; ++r)
        Pt[(wv * 16 + r + 8 * half) * PS + f * 16 + l16] = f2bf(sf[f][r]);

    // [B] all waves done reading Kt -> safe to start K(j+1); V(j) must land
    __syncthreads();
    if (wv == 0) {
      if (j < qb) {
        tdm_load_2d(kt_lds, Kb + (size_t)((j + 1) * 128) * KQW, 128, 128, KQW, 5, 3);
        __builtin_amdgcn_s_wait_tensorcnt(1);   // in-order: V(j) done, K(j+1) in flight
      } else {
        __builtin_amdgcn_s_wait_tensorcnt(0);   // last block: just V(j)
      }
    }
    __syncthreads();   // [C] V^T visible to all waves

    // O += P * V : 4 key-chunks x 8 hd-subtiles = 32 WMMA per wave
#pragma unroll
    for (int kc = 0; kc < 4; ++kc) {
      AFrag pf;
      const unsigned short* pp = &Pt[(wv * 16 + l16) * PS];
#pragma unroll
      for (int v = 0; v < 8; ++v)
        pf.d[v] = *(const unsigned int*)&pp[kc * 32 + (v >= 4 ? 16 : 0) + half * 8 + (v & 3) * 2];
#pragma unroll
      for (int f = 0; f < 8; ++f) {
        AFrag vf;
        const unsigned short* vp = &Vb[(f * 16 + l16) * PS + kc * 32];
#pragma unroll
        for (int v = 0; v < 8; ++v) vf.d[v] = *(const unsigned int*)&vp[(v >= 4 ? 16 : 0) + half * 8 + (v & 3) * 2];
        o[f] = __builtin_amdgcn_wmma_f32_16x16x32_bf16(
            false, pf.v, false, vf.v, (short)0, o[f], false, false);
      }
    }
  }

  // normalize and store out[b][h][sq][d] (f32)
  float inv[8];
#pragma unroll
  for (int r = 0; r < 8; ++r) inv[r] = 1.0f / lrun[r];
  float* ob = out + (size_t)(b * 16 + h) * S * 128;
#pragma unroll
  for (int f = 0; f < 8; ++f) {
    int d = f * 16 + l16;
#pragma unroll
    for (int r = 0; r < 8; ++r) {
      int sq = q0 + r + 8 * half;
      ob[(size_t)sq * 128 + d] = o[f][r] * inv[r];
    }
  }
}

// ---------------------------------------------------------------------------
extern "C" void kernel_launch(void* const* d_in, const int* in_sizes, int n_in,
                              void* d_out, int out_size, void* d_ws, size_t ws_size,
                              hipStream_t stream) {
  const float* x  = (const float*)d_in[0];   // [2,2048,2048]
  const float* Wk = (const float*)d_in[1];   // [2048,6144]
  const float* bk = (const float*)d_in[2];   // [6144]
  float* out      = (float*)d_out;           // [2,16,2048,128]

  char* ws = (char*)d_ws;                    // layout (bytes):
  unsigned short* xbf = (unsigned short*)(ws);                             // 16 MB
  unsigned short* wT  = (unsigned short*)(ws + (size_t)16 * 1024 * 1024);  // 24 MB
  unsigned short* kq  = (unsigned short*)(ws + (size_t)40 * 1024 * 1024);  // 32 MB
  unsigned short* vt  = (unsigned short*)(ws + (size_t)72 * 1024 * 1024);  // 16 MB

  cvt_x_bf16<<<dim3((GM * GK) / (256 * 8)), 256, 0, stream>>>(x, xbf);
  cvt_wT_bf16<<<dim3(GN / 64, GK / 64), 256, 0, stream>>>(Wk, wT);
  kqv_gemm_bf16<<<dim3(GN / 128, GM / 128), 256, 0, stream>>>(xbf, wT, bk, kq, vt);
  alibi_fattn<<<512, 256, 0, stream>>>(kq, vt, out);
}
